// GCEA_Block_10075993276883
// MI455X (gfx1250) — matrix operand order repaired and checked by reading, not compiled, not run on hardware
//
#include <hip/hip_runtime.h>

// ---------------------------------------------------------------------------
// GCEA block for MI455X (gfx1250, wave32, WMMA).
// B=16, C=256, H=W=64, HW=4096, K(attn)=64.
// Pipeline: stats -> dvec -> mask logits -> softmax -> context pooling ->
// fused external-attention GEMM/softmax/GEMM with v_wmma_f32_16x16x32_f16.
// ---------------------------------------------------------------------------

#define Bn   16
#define Cn   256
#define HWn  4096
#define Kn   64

typedef __attribute__((ext_vector_type(16))) _Float16 v16h;
typedef __attribute__((ext_vector_type(8)))  float    v8f;

#define WMMA_F16(A, Bm, Cc) \
  __builtin_amdgcn_wmma_f32_16x16x32_f16(false, (A), false, (Bm), (short)0, (Cc), false, false)

// A-operand fragment (16x32 f16, ISA 7.12.2): lane l: row = l&15, h = l>>4.
// VGPR p holds K = k0 + 16*(p>>2) + 8*h + 2*(p&3)  and K+1 (packed pair).
__device__ __forceinline__ v16h fragA_ld(const _Float16* row, int k0, int h) {
  union { v16h v; unsigned u[8]; } r;
#pragma unroll
  for (int p = 0; p < 8; ++p) {
    int K = k0 + ((p >> 2) << 4) + (h << 3) + ((p & 3) << 1);
    r.u[p] = *(const unsigned*)(row + K);
  }
  return r.v;
}

// B-operand fragment (32x16 f16): lane l: col = l&15, h = l>>4.
// VGPR p holds K = k0 + 16*(p>>2) + 8*h + 2*(p&3) and K+1. `row` points at column data laid
// out contiguously in K (we store W^T rows in LDS).
__device__ __forceinline__ v16h fragB_ld(const _Float16* row, int k0, int h) {
  union { v16h v; unsigned u[8]; } r;
#pragma unroll
  for (int p = 0; p < 8; ++p) {
    int K = k0 + ((p >> 2) << 4) + (h << 3) + ((p & 3) << 1);
    r.u[p] = *(const unsigned*)(row + K);
  }
  return r.v;
}

// --------------------------- kernel 1: stats -------------------------------
// alpha[b,c] = w[c] * rsqrt(var+eps); muw[b,c] = mu * alpha  (for d[b])
__global__ __launch_bounds__(256) void k_stats(const float* __restrict__ style,
                                               const float* __restrict__ w,
                                               float* __restrict__ alpha,
                                               float* __restrict__ muw) {
  int bc = blockIdx.x;                       // b*C + c
  const float* p = style + (size_t)bc * HWn;
  float s = 0.f, ss = 0.f;
  for (int i = threadIdx.x; i < HWn; i += 256) { float v = p[i]; s += v; ss += v * v; }
  __shared__ float S[256], SS[256];
  S[threadIdx.x] = s; SS[threadIdx.x] = ss;
  __syncthreads();
  for (int o = 128; o > 0; o >>= 1) {
    if (threadIdx.x < o) { S[threadIdx.x] += S[threadIdx.x + o]; SS[threadIdx.x] += SS[threadIdx.x + o]; }
    __syncthreads();
  }
  if (threadIdx.x == 0) {
    float mu  = S[0] * (1.f / HWn);
    float var = SS[0] * (1.f / HWn) - mu * mu;
    float rs  = rsqrtf(var + 1e-5f);
    float a   = w[bc & (Cn - 1)] * rs;
    alpha[bc] = a;
    muw[bc]   = mu * a;
  }
}

// --------------------------- kernel 2: d[b] --------------------------------
__global__ __launch_bounds__(256) void k_dvec(const float* __restrict__ muw,
                                              const float* __restrict__ bconst,
                                              float* __restrict__ dvec) {
  __shared__ float S[256];
  int t = threadIdx.x;
  S[t] = muw[blockIdx.x * Cn + t];
  __syncthreads();
  for (int o = 128; o > 0; o >>= 1) {
    if (t < o) S[t] += S[t + o];
    __syncthreads();
  }
  if (t == 0) dvec[blockIdx.x] = bconst[0] - S[0];
}

// --------------------------- kernel 3: mask logits -------------------------
// logits[b,hw] = sum_c style[b,c,hw]*alpha[b,c] + d[b]
__global__ __launch_bounds__(256) void k_logits(const float* __restrict__ style,
                                                const float* __restrict__ alpha,
                                                const float* __restrict__ dvec,
                                                float* __restrict__ logits) {
  int b  = blockIdx.x >> 4;
  int hw = ((blockIdx.x & 15) << 8) + threadIdx.x;
  __shared__ float aL[Cn];
  aL[threadIdx.x] = alpha[b * Cn + threadIdx.x];
  __syncthreads();
  const float* p = style + ((size_t)b * Cn) * HWn + hw;
  float acc = dvec[b];
#pragma unroll 4
  for (int c = 0; c < Cn; ++c) acc += p[(size_t)c * HWn] * aL[c];
  logits[b * HWn + hw] = acc;
}

// --------------------------- kernel 4: spatial softmax ---------------------
__global__ __launch_bounds__(256) void k_softmax(const float* __restrict__ logits,
                                                 float* __restrict__ maskv) {
  __shared__ float S[256];
  int b = blockIdx.x, t = threadIdx.x;
  const float* p = logits + b * HWn;
  float v[16];
  float m = -3.4e38f;
#pragma unroll
  for (int i = 0; i < 16; ++i) { v[i] = p[t + (i << 8)]; m = fmaxf(m, v[i]); }
  S[t] = m; __syncthreads();
  for (int o = 128; o > 0; o >>= 1) { if (t < o) S[t] = fmaxf(S[t], S[t + o]); __syncthreads(); }
  float M = S[0]; __syncthreads();
  float s = 0.f;
#pragma unroll
  for (int i = 0; i < 16; ++i) { v[i] = __expf(v[i] - M); s += v[i]; }
  S[t] = s; __syncthreads();
  for (int o = 128; o > 0; o >>= 1) { if (t < o) S[t] += S[t + o]; __syncthreads(); }
  float inv = 1.0f / S[0];
  float* q = maskv + b * HWn;
#pragma unroll
  for (int i = 0; i < 16; ++i) q[t + (i << 8)] = v[i] * inv;
}

// --------------------------- kernel 5: context pooling ---------------------
__global__ __launch_bounds__(256) void k_context(const float* __restrict__ style,
                                                 const float* __restrict__ maskv,
                                                 float* __restrict__ ctxv) {
  int bc = blockIdx.x;
  const float* p  = style + (size_t)bc * HWn;
  const float* mk = maskv + (bc >> 8) * HWn;
  float s = 0.f;
  for (int i = threadIdx.x; i < HWn; i += 256) s += p[i] * mk[i];
  __shared__ float S[256];
  S[threadIdx.x] = s; __syncthreads();
  for (int o = 128; o > 0; o >>= 1) {
    if (threadIdx.x < o) S[threadIdx.x] += S[threadIdx.x + o];
    __syncthreads();
  }
  if (threadIdx.x == 0) ctxv[bc] = S[0];
}

// --------------------------- kernel 6: fused external attention ------------
// 4 waves/block, one 16-row WMMA tile per wave; 64 rows of n per block.
#define NW 4
#define MK_STRIDE 264   // 256 + 8 pad (bank-conflict avoidance)
#define CT_STRIDE 264
#define MV_STRIDE 72    // 64 + 8 pad
#define AT_STRIDE 72

__global__ __launch_bounds__(32 * NW) void k_extattn(const float* __restrict__ content,
                                                     const float* __restrict__ ctxv,
                                                     const float* __restrict__ mkw,
                                                     const float* __restrict__ mvw,
                                                     float* __restrict__ out) {
  __shared__ _Float16 sMK[Kn * MK_STRIDE];       // mk_w[k][c] as f16
  __shared__ _Float16 sMV[Cn * MV_STRIDE];       // mv_w[c][k] as f16
  __shared__ _Float16 sCT[NW][16 * CT_STRIDE];   // per-wave ctx tile [row][c]
  __shared__ _Float16 sAT[NW][16 * AT_STRIDE];   // per-wave attn tile [row][k]

  int tid  = threadIdx.x;
  int wave = tid >> 5, lane = tid & 31;
  int b    = blockIdx.x >> 6;                    // 64 blocks per batch
  int n0   = (blockIdx.x & 63) * (16 * NW);

  // stage weights (f32 -> f16) cooperatively
  for (int i = tid; i < Kn * Cn; i += 32 * NW) {
    int k = i >> 8, c = i & (Cn - 1);
    sMK[k * MK_STRIDE + c] = (_Float16)mkw[i];
  }
  for (int i = tid; i < Cn * Kn; i += 32 * NW) {
    int c = i >> 6, k = i & (Kn - 1);
    sMV[c * MV_STRIDE + k] = (_Float16)mvw[i];
  }

  // stage this wave's ctx tile: ctx[b,n,c] = content_flat + context[b][n>>4]
  int nbase   = n0 + wave * 16;
  float addv  = ctxv[b * Cn + (nbase >> 4)];     // constant per 16-row tile
  const float* cptr = content + ((size_t)b * HWn + nbase) * Cn;
  for (int i = lane; i < 16 * Cn / 4; i += 32) {
    int e = i << 2, row = e >> 8, col = e & (Cn - 1);
    float4 f = *(const float4*)(cptr + row * Cn + col);
    _Float16* dst = &sCT[wave][row * CT_STRIDE + col];
    dst[0] = (_Float16)(f.x + addv); dst[1] = (_Float16)(f.y + addv);
    dst[2] = (_Float16)(f.z + addv); dst[3] = (_Float16)(f.w + addv);
  }
  __syncthreads();

  int r = lane & 15, h = lane >> 4;
  const _Float16* aRow = &sCT[wave][r * CT_STRIDE];

  // GEMM1: attn[16,64] = ctx[16,256] x mk_w^T[256,64], K in 8 steps of 32
  v8f A0 = {}, A1 = {}, A2 = {}, A3 = {};
#pragma unroll
  for (int kc = 0; kc < 8; ++kc) {
    int k0 = kc * 32;
    v16h af = fragA_ld(aRow, k0, h);
    A0 = WMMA_F16(af, fragB_ld(&sMK[(0 * 16 + r) * MK_STRIDE], k0, h), A0);
    A1 = WMMA_F16(af, fragB_ld(&sMK[(1 * 16 + r) * MK_STRIDE], k0, h), A1);
    A2 = WMMA_F16(af, fragB_ld(&sMK[(2 * 16 + r) * MK_STRIDE], k0, h), A2);
    A3 = WMMA_F16(af, fragB_ld(&sMK[(3 * 16 + r) * MK_STRIDE], k0, h), A3);
  }

  // softmax over k=64 per row + faithful renorm; row m=v+8h lives in one
  // 16-lane group (one column per lane, 4 tile-slots) -> shfl_xor reductions
#pragma unroll
  for (int v = 0; v < 8; ++v) {
    float x0 = A0[v], x1 = A1[v], x2 = A2[v], x3 = A3[v];
    float m = fmaxf(fmaxf(x0, x1), fmaxf(x2, x3));
    for (int d = 1; d < 16; d <<= 1) m = fmaxf(m, __shfl_xor(m, d));
    float e0 = __expf(x0 - m), e1 = __expf(x1 - m),
          e2 = __expf(x2 - m), e3 = __expf(x3 - m);
    float s = e0 + e1 + e2 + e3;
    for (int d = 1; d < 16; d <<= 1) s += __shfl_xor(s, d);
    float is = 1.0f / s;
    float r0 = e0 * is, r1 = e1 * is, r2 = e2 * is, r3 = e3 * is;
    float s2 = r0 + r1 + r2 + r3;
    for (int d = 1; d < 16; d <<= 1) s2 += __shfl_xor(s2, d);
    float q = 1.0f / s2;
    _Float16* arow = &sAT[wave][(v + 8 * h) * AT_STRIDE];
    arow[0 * 16 + r] = (_Float16)(r0 * q);
    arow[1 * 16 + r] = (_Float16)(r1 * q);
    arow[2 * 16 + r] = (_Float16)(r2 * q);
    arow[3 * 16 + r] = (_Float16)(r3 * q);
  }
  // per-wave LDS tile: same-wave DS ops are in-order, no block barrier needed

  // GEMM2: out[16,256] = attn[16,64] x mv_w^T[64,256], K in 2 steps of 32
  v16h P0 = fragA_ld(&sAT[wave][r * AT_STRIDE], 0,  h);
  v16h P1 = fragA_ld(&sAT[wave][r * AT_STRIDE], 32, h);
  float* obase = out + ((size_t)b * HWn + nbase) * Cn;
#pragma unroll
  for (int t = 0; t < 16; ++t) {
    v8f acc = {};
    acc = WMMA_F16(P0, fragB_ld(&sMV[(t * 16 + r) * MV_STRIDE], 0,  h), acc);
    acc = WMMA_F16(P1, fragB_ld(&sMV[(t * 16 + r) * MV_STRIDE], 32, h), acc);
    float* oc = obase + t * 16 + r;           // raw reshape: out[b, n, c] flat
#pragma unroll
    for (int v = 0; v < 8; ++v) oc[(size_t)(v + 8 * h) * Cn] = acc[v];
  }
}

// ---------------------------------------------------------------------------
extern "C" void kernel_launch(void* const* d_in, const int* in_sizes, int n_in,
                              void* d_out, int out_size, void* d_ws, size_t ws_size,
                              hipStream_t stream) {
  (void)in_sizes; (void)n_in; (void)out_size; (void)ws_size;
  const float* content = (const float*)d_in[0];
  const float* style   = (const float*)d_in[1];
  const float* cmask_w = (const float*)d_in[2];
  const float* cmask_b = (const float*)d_in[3];
  const float* mk_w    = (const float*)d_in[4];
  const float* mv_w    = (const float*)d_in[5];
  float* out = (float*)d_out;

  float* ws     = (float*)d_ws;
  float* alpha  = ws;                                  // B*C      = 4096
  float* muw    = alpha + Bn * Cn;                     // B*C      = 4096
  float* dvec   = muw + Bn * Cn;                       // B        = 16
  float* logits = dvec + Bn;                           // B*HW     = 65536
  float* maskv  = logits + Bn * HWn;                   // B*HW     = 65536
  float* ctxv   = maskv + Bn * HWn;                    // B*C      = 4096

  k_stats  <<<Bn * Cn, 256, 0, stream>>>(style, cmask_w, alpha, muw);
  k_dvec   <<<Bn,      256, 0, stream>>>(muw, cmask_b, dvec);
  k_logits <<<Bn * 16, 256, 0, stream>>>(style, alpha, dvec, logits);
  k_softmax<<<Bn,      256, 0, stream>>>(logits, maskv);
  k_context<<<Bn * Cn, 256, 0, stream>>>(style, maskv, ctxv);
  k_extattn<<<Bn * 64, 32 * NW, 0, stream>>>(content, ctxv, mk_w, mv_w, out);
}
